// myGCNmodule_28819230556648
// MI455X (gfx1250) — matrix operand order repaired and checked by reading, not compiled, not run on hardware
//
#include <hip/hip_runtime.h>
#include <hip/hip_bf16.h>
#include <math.h>

#define B_   4
#define N_   8192
#define D_   8
#define E_   128
#define S_   9
#define BN   (B_ * N_)        // 32768 rows
#define NCOL 768              // 2 dirs * 3 mats * 128
#define SCALE_F 0.08838834764831845f  // 1/sqrt(128)

typedef __bf16 bf16;
typedef __attribute__((ext_vector_type(8)))  __bf16 v8bf;
typedef __attribute__((ext_vector_type(16))) __bf16 v16bf;
typedef __attribute__((ext_vector_type(8)))  float  v8f;

// Build a v16bf from two 8-element contiguous chunks (16B loads).
__device__ __forceinline__ v16bf ld16(const bf16* p0, const bf16* p1) {
    v8bf a0 = *reinterpret_cast<const v8bf*>(p0);
    v8bf a1 = *reinterpret_cast<const v8bf*>(p1);
    v16bf a;
#pragma unroll
    for (int i = 0; i < 8; ++i) { a[i] = a0[i]; a[i + 8] = a1[i]; }
    return a;
}

// ---------------------------------------------------------------- X -> bf16
__global__ void k_cvt(const float* __restrict__ X, bf16* __restrict__ Xh, int n) {
    int t = blockIdx.x * blockDim.x + threadIdx.x;
    if (t < n) Xh[t] = (bf16)X[t];
}

// ------------------------------------------- fold weights into WMMA layouts
// Bt[ng][k]  (ng = dir*384 + mat*128 + n) : Bt[ng][k] = sum_j Wm[k][j]*ipw[mat*128+n][j]
// G[dir][n][k] = sum_j lin_w[n][dir*128+j] * ow_dir[j][k]
// biasc[ng] = ipb[mat*128+n];  btot[n] = lin_b[n] + lw0[n]·ob_in + lw1[n]·ob_out
__global__ void k_prep(const float* __restrict__ Wq_i, const float* __restrict__ Wk_i,
                       const float* __restrict__ Wv_i, const float* __restrict__ ipw_i,
                       const float* __restrict__ ipb_i, const float* __restrict__ ow_i,
                       const float* __restrict__ ob_i,
                       const float* __restrict__ Wq_o, const float* __restrict__ Wk_o,
                       const float* __restrict__ Wv_o, const float* __restrict__ ipw_o,
                       const float* __restrict__ ipb_o, const float* __restrict__ ow_o,
                       const float* __restrict__ ob_o,
                       const float* __restrict__ lin_w, const float* __restrict__ lin_b,
                       bf16* __restrict__ Bt, float* __restrict__ biasc,
                       bf16* __restrict__ G, float* __restrict__ btot) {
    int t = blockIdx.x * blockDim.x + threadIdx.x;   // exactly 131072 threads
    if (t < NCOL * E_) {
        int ng = t >> 7, k = t & 127;
        int dir = ng / 384, rem = ng % 384, mat = rem >> 7, n = rem & 127;
        const float* Wm = dir ? (mat == 0 ? Wq_o : (mat == 1 ? Wk_o : Wv_o))
                              : (mat == 0 ? Wq_i : (mat == 1 ? Wk_i : Wv_i));
        const float* ipw  = dir ? ipw_o : ipw_i;
        const float* wrow = ipw + (mat * E_ + n) * E_;
        const float* Wrow = Wm + k * E_;
        float acc = 0.f;
        for (int j = 0; j < E_; ++j) acc += Wrow[j] * wrow[j];
        Bt[ng * E_ + k] = (bf16)acc;
        if (k == 0) {
            const float* ipb = dir ? ipb_o : ipb_i;
            biasc[ng] = ipb[mat * E_ + n];
        }
    } else {
        int e = t - NCOL * E_;                       // < 32768
        int dir = e >> 14, rem = e & 16383, n = rem >> 7, k = rem & 127;
        const float* ow = dir ? ow_o : ow_i;
        float acc = 0.f;
        for (int j = 0; j < E_; ++j) acc += lin_w[n * 2 * E_ + dir * E_ + j] * ow[j * E_ + k];
        G[(dir * E_ + n) * E_ + k] = (bf16)acc;
        if (dir == 0 && k == 0) {
            float s = lin_b[n];
            for (int j = 0; j < E_; ++j)
                s += lin_w[n * 2 * E_ + j] * ob_i[j] + lin_w[n * 2 * E_ + E_ + j] * ob_o[j];
            btot[n] = s;
        }
    }
}

// ------------------------------------------------ QKV GEMM: Y = Xh @ Bt^T + b
// M=32768, N=768, K=128. Each wave computes a 32x64 block (2 m-tiles x 4 n-tiles):
// each A fragment feeds 4 WMMAs, each B fragment feeds 2 -> balanced L2 traffic.
__global__ void k_qkv(const bf16* __restrict__ Xh, const bf16* __restrict__ Bt,
                      const float* __restrict__ biasc, bf16* __restrict__ Y) {
    const int NG = NCOL / 64;                        // 12 n-groups of 4 tiles
    int wave = (blockIdx.x * blockDim.x + threadIdx.x) >> 5;
    int lane = threadIdx.x & 31;
    int mg = wave / NG, ng = wave % NG;              // mg: 32-row group
    int half = lane >> 4, r = lane & 15;
    v8f acc[2][4] = {{{}, {}, {}, {}}, {{}, {}, {}, {}}};
#pragma unroll
    for (int kc = 0; kc < 4; ++kc) {
        int k0 = kc * 32;
        // A 16x32 bf16: lane<16 holds row r, K {k0..k0+7, k0+16..k0+23}; lane>=16 the other half
        const bf16* pa0 = Xh + (size_t)(mg * 32 + r) * E_ + k0 + half * 8;
        const bf16* pa1 = pa0 + (size_t)16 * E_;
        v16bf a0 = ld16(pa0, pa0 + 16);
        v16bf a1 = ld16(pa1, pa1 + 16);
        // B 32x16 bf16: lane = half*16 + n holds column n, K {k0+half*8..+7, k0+16+half*8..+7}
        v16bf bb[4];
#pragma unroll
        for (int t = 0; t < 4; ++t) {
            const bf16* pb = Bt + (size_t)((ng * 4 + t) * 16 + r) * E_ + k0 + half * 8;
            bb[t] = ld16(pb, pb + 16);
        }
#pragma unroll
        for (int t = 0; t < 4; ++t)
            acc[0][t] = __builtin_amdgcn_wmma_f32_16x16x32_bf16(false, a0, false, bb[t],
                                                                (short)0, acc[0][t], false, false);
#pragma unroll
        for (int t = 0; t < 4; ++t)
            acc[1][t] = __builtin_amdgcn_wmma_f32_16x16x32_bf16(false, a1, false, bb[t],
                                                                (short)0, acc[1][t], false, false);
    }
#pragma unroll
    for (int mt = 0; mt < 2; ++mt) {
#pragma unroll
        for (int t = 0; t < 4; ++t) {
            int col = (ng * 4 + t) * 16 + r;
            float bias = biasc[col];
#pragma unroll
            for (int r8 = 0; r8 < 8; ++r8) {
                int row = mg * 32 + mt * 16 + half * 8 + r8;  // D: M = half*8+r8, N = lane&15
                Y[(size_t)row * NCOL + col] = (bf16)(acc[mt][t][r8] + bias);
            }
        }
    }
}

// ----------------------------------- attention: one wave per (node, direction)
__global__ void k_attn(const bf16* __restrict__ Y,
                       const int* __restrict__ in_idx, const int* __restrict__ out_idx,
                       const int* __restrict__ in_mask, const int* __restrict__ out_mask,
                       bf16* __restrict__ ctx) {
    int wave = (blockIdx.x * blockDim.x + threadIdx.x) >> 5;
    int lane = threadIdx.x & 31;
    int dir  = wave >> 15;                            // 32768 nodes per direction
    int node = wave & (BN - 1);
    int bbase = node & ~(N_ - 1);                     // start row of this batch
    const int* idx = dir ? out_idx : in_idx;
    const int* msk = dir ? out_mask : in_mask;
    const bf16* Yq = Y + (size_t)node * NCOL + dir * 384;

    float q[4];
#pragma unroll
    for (int j = 0; j < 4; ++j) q[j] = (float)Yq[lane * 4 + j];

    int  rows[S_]; bool valid[S_];
    rows[0] = node; valid[0] = true;
#pragma unroll
    for (int s = 1; s < S_; ++s) {
        rows[s]  = bbase + idx[node * D_ + s - 1];
        valid[s] = (msk[node * D_ + s - 1] != 0);
    }

    float sc[S_];
#pragma unroll
    for (int s = 0; s < S_; ++s) {
        const bf16* kr = Y + (size_t)rows[s] * NCOL + dir * 384 + E_ + lane * 4;
        float d = 0.f;
#pragma unroll
        for (int j = 0; j < 4; ++j) d += q[j] * (float)kr[j];
#pragma unroll
        for (int off = 16; off > 0; off >>= 1) d += __shfl_xor(d, off, 32);
        sc[s] = valid[s] ? d * SCALE_F : -3.40282347e38f;
    }
    float mx = sc[0];
#pragma unroll
    for (int s = 1; s < S_; ++s) mx = fmaxf(mx, sc[s]);
    float w[S_], sum = 0.f;
#pragma unroll
    for (int s = 0; s < S_; ++s) { w[s] = __expf(sc[s] - mx); sum += w[s]; }
    float inv = 1.f / sum;

    float c[4] = {0.f, 0.f, 0.f, 0.f};
#pragma unroll
    for (int s = 0; s < S_; ++s) {
        const bf16* vr = Y + (size_t)rows[s] * NCOL + dir * 384 + 2 * E_ + lane * 4;
        float ws = w[s] * inv;
#pragma unroll
        for (int j = 0; j < 4; ++j) c[j] += ws * (float)vr[j];
    }
    bf16* co = ctx + ((size_t)(dir * BN + node)) * E_ + lane * 4;
#pragma unroll
    for (int j = 0; j < 4; ++j) co[j] = (bf16)c[j];
}

// ---------------- fused out_proj + final linear + ELU: out = elu(ctx·Gᵀ + b)
// Each wave computes a 32x64 block (2 m-tiles x 4 n-tiles), both directions
// accumulated into the same tiles.
__global__ void k_out(const bf16* __restrict__ ctx, const bf16* __restrict__ G,
                      const float* __restrict__ btot, float* __restrict__ out) {
    const int NG = E_ / 64;                          // 2 n-groups of 4 tiles
    int wave = (blockIdx.x * blockDim.x + threadIdx.x) >> 5;
    int lane = threadIdx.x & 31;
    int mg = wave / NG, ng = wave % NG;
    int half = lane >> 4, r = lane & 15;
    v8f acc[2][4] = {{{}, {}, {}, {}}, {{}, {}, {}, {}}};
#pragma unroll
    for (int dir = 0; dir < 2; ++dir) {
        const bf16* A  = ctx + (size_t)dir * BN * E_;
        const bf16* Bm = G + dir * E_ * E_;
#pragma unroll
        for (int kc = 0; kc < 4; ++kc) {
            int k0 = kc * 32;
            const bf16* pa0 = A + (size_t)(mg * 32 + r) * E_ + k0 + half * 8;
            const bf16* pa1 = pa0 + (size_t)16 * E_;
            v16bf a0 = ld16(pa0, pa0 + 16);
            v16bf a1 = ld16(pa1, pa1 + 16);
            v16bf bb[4];
#pragma unroll
            for (int t = 0; t < 4; ++t) {
                const bf16* pb = Bm + (size_t)((ng * 4 + t) * 16 + r) * E_ + k0 + half * 8;
                bb[t] = ld16(pb, pb + 16);
            }
#pragma unroll
            for (int t = 0; t < 4; ++t)
                acc[0][t] = __builtin_amdgcn_wmma_f32_16x16x32_bf16(false, a0, false, bb[t],
                                                                    (short)0, acc[0][t], false, false);
#pragma unroll
            for (int t = 0; t < 4; ++t)
                acc[1][t] = __builtin_amdgcn_wmma_f32_16x16x32_bf16(false, a1, false, bb[t],
                                                                    (short)0, acc[1][t], false, false);
        }
    }
#pragma unroll
    for (int mt = 0; mt < 2; ++mt) {
#pragma unroll
        for (int t = 0; t < 4; ++t) {
            int col = (ng * 4 + t) * 16 + r;
            float bias = btot[col];
#pragma unroll
            for (int r8 = 0; r8 < 8; ++r8) {
                int row = mg * 32 + mt * 16 + half * 8 + r8;
                float v = acc[mt][t][r8] + bias;
                out[(size_t)row * E_ + col] = (v > 0.f) ? v : (__expf(v) - 1.f);
            }
        }
    }
}

extern "C" void kernel_launch(void* const* d_in, const int* in_sizes, int n_in,
                              void* d_out, int out_size, void* d_ws, size_t ws_size,
                              hipStream_t stream) {
    const float* X        = (const float*)d_in[0];
    const int*   in_idx   = (const int*)d_in[1];
    const int*   out_idx  = (const int*)d_in[2];
    const int*   in_mask  = (const int*)d_in[3];
    const int*   out_mask = (const int*)d_in[4];
    const float* in_Wq  = (const float*)d_in[5];
    const float* in_Wk  = (const float*)d_in[6];
    const float* in_Wv  = (const float*)d_in[7];
    const float* in_ipw = (const float*)d_in[8];
    const float* in_ipb = (const float*)d_in[9];
    const float* in_ow  = (const float*)d_in[10];
    const float* in_ob  = (const float*)d_in[11];
    const float* out_Wq  = (const float*)d_in[12];
    const float* out_Wk  = (const float*)d_in[13];
    const float* out_Wv  = (const float*)d_in[14];
    const float* out_ipw = (const float*)d_in[15];
    const float* out_ipb = (const float*)d_in[16];
    const float* out_ow  = (const float*)d_in[17];
    const float* out_ob  = (const float*)d_in[18];
    const float* lin_w   = (const float*)d_in[19];
    const float* lin_b   = (const float*)d_in[20];
    float* out = (float*)d_out;

    // workspace layout (~72.3 MB)
    bf16*  Xh    = (bf16*)d_ws;                       //  8 MB
    bf16*  Y     = Xh  + (size_t)BN * E_;             // 48 MB (QKV both dirs, bf16)
    bf16*  ctx   = Y   + (size_t)BN * NCOL;           // 16 MB
    bf16*  Bt    = ctx + (size_t)2 * BN * E_;         // 192 KB
    bf16*  G     = Bt  + (size_t)NCOL * E_;           // 64 KB
    float* biasc = (float*)(G + (size_t)2 * E_ * E_); //  3 KB
    float* btot  = biasc + NCOL;                      // 512 B

    k_cvt<<<(BN * E_ + 255) / 256, 256, 0, stream>>>(X, Xh, BN * E_);
    k_prep<<<512, 256, 0, stream>>>(in_Wq, in_Wk, in_Wv, in_ipw, in_ipb, in_ow, in_ob,
                                    out_Wq, out_Wk, out_Wv, out_ipw, out_ipb, out_ow, out_ob,
                                    lin_w, lin_b, Bt, biasc, G, btot);
    // 1024 m-groups (32 rows) * 12 n-groups (64 cols), 4 waves/block
    k_qkv<<<(1024 * 12) / 4, 128, 0, stream>>>(Xh, Bt, biasc, Y);
    // 65536 waves (node,dir), 4 waves/block
    k_attn<<<(2 * BN) / 4, 128, 0, stream>>>(Y, in_idx, out_idx, in_mask, out_mask, ctx);
    // 1024 m-groups * 2 n-groups, 4 waves/block
    k_out<<<(1024 * 2) / 4, 128, 0, stream>>>(ctx, G, btot, out);
}
